// SelfMultiheadAttention_63213328663277
// MI455X (gfx1250) — compile-verified, matrix operand order checked by reading, not run on hardware
//
#include <hip/hip_runtime.h>
#include <hip/hip_bf16.h>
#include <math.h>

// ---------------------------------------------------------------------------
// MI455X (gfx1250) self-attention with "unknown" stream.
// bf16 operands + fp32 accumulation via v_wmma_f32_16x16x32_bf16.
//   - weights pre-transposed once (Wt[n][k]) -> GEMM B-fragments contiguous
//   - V projection stored transposed (Vt[bh][d][c]) -> P.V B-fragments contiguous
//   - attention K / Vt tiles staged into LDS with double-buffered TDM DMA
//     (tensor_load_to_lds + s_wait_tensorcnt)
//   - softmax uses lane-contiguous 32-element chunks -> b128 LDS traffic
// ---------------------------------------------------------------------------

typedef __bf16 bf16;
typedef __attribute__((ext_vector_type(16))) __bf16 v16bf;
typedef __attribute__((ext_vector_type(8)))  __bf16 v8bf;
typedef __attribute__((ext_vector_type(8)))  float  v8f;
typedef __attribute__((ext_vector_type(4)))  unsigned int v4u;
typedef __attribute__((ext_vector_type(8)))  int v8i;
typedef __attribute__((ext_vector_type(4)))  int v4i;

#define B_  4
#define C_  1024
#define E_  1024
#define H_  16
#define D_  64
#define SCALE_ 0.125f   // 1/sqrt(64)

// LDS layout inside smha_attn (dynamic shared, base offset 0):
//   [0      , 32768)  TDM staging buffer 0
//   [32768  , 65536)  TDM staging buffer 1
//   [65536  , 131072) fp32 score buffer (16 x 1024), later P1|P2 bf16 in place
#define LDS_BUF0 0u
#define LDS_BUF1 32768u
#define LDS_SC   65536u
#define LDS_TOTAL 131072u

// ---------------- wave32 reductions ----------------
__device__ inline float wred_max(float x) {
#pragma unroll
  for (int o = 16; o > 0; o >>= 1) x = fmaxf(x, __shfl_xor(x, o, 32));
  return x;
}
__device__ inline float wred_sum(float x) {
#pragma unroll
  for (int o = 16; o > 0; o >>= 1) x += __shfl_xor(x, o, 32);
  return x;
}

// ---------------- CDNA5 16-bit WMMA fragment packing -----------------------
// A (16x32): lane L holds row M=L%16; lanes 0-15 get K{0-7,16-23},
// lanes 16-31 get K{8-15,24-31}. B (32x16) mirrored with N=L%16.
__device__ inline v16bf frag_contig(const bf16* base, int koff) {
  v8bf lo = *(const v8bf*)(base + koff);
  v8bf hi = *(const v8bf*)(base + koff + 16);
  return __builtin_shufflevector(lo, hi, 0, 1, 2, 3, 4, 5, 6, 7,
                                 8, 9, 10, 11, 12, 13, 14, 15);
}
__device__ inline v8f wmma_bf16(v16bf a, v16bf b, v8f c) {
  return __builtin_amdgcn_wmma_f32_16x16x32_bf16(false, a, false, b, (short)0, c, false, false);
}

// ---------------- Tensor Data Mover: 2D bf16 tile -> LDS -------------------
// D# per CDNA5 ISA.
// group0: [count=1 | lds_addr | global_addr(57b) | type=2]
// group1: data_size=2B, tensor_dim0/1, tile_dim0/1, tensor_dim0_stride.
// Dims in elements; groups 2/3 NULL-equivalent (zero) for 2D tensors.
__device__ inline void tdm_load_2d_bf16(unsigned lds_off, const void* gaddr,
                                        int tile_w, int tile_h,
                                        int tens_w, int tens_h, int row_stride) {
  unsigned long long ga = (unsigned long long)gaddr;
  v4u g0;
  g0[0] = 1u;                                        // count=1, user mode
  g0[1] = lds_off;                                   // lds_addr (bytes)
  g0[2] = (unsigned)ga;                              // global_addr[31:0]
  g0[3] = (unsigned)(ga >> 32) | (2u << 30);         // global_addr[56:32] | type=2
  v8i g1;
  g1[0] = 1 << 16;                                   // data_size=1 (2 bytes)
  g1[1] = (tens_w & 0xffff) << 16;                   // tensor_dim0[15:0]
  g1[2] = ((tens_w >> 16) & 0xffff) | ((tens_h & 0xffff) << 16);
  g1[3] = ((tens_h >> 16) & 0xffff) | ((tile_w & 0xffff) << 16);  // tile_dim0
  g1[4] = tile_h & 0xffff;                           // tile_dim1 (tile_dim2=0)
  g1[5] = row_stride;                                // tensor_dim0_stride[31:0]
  g1[6] = 0;
  g1[7] = 0;
  v4i z4 = {0, 0, 0, 0};
#if __clang_major__ >= 23
  v8i z8 = {0, 0, 0, 0, 0, 0, 0, 0};
  __builtin_amdgcn_tensor_load_to_lds(g0, g1, z4, z4, z8, 0);
#else
  __builtin_amdgcn_tensor_load_to_lds(g0, g1, z4, z4, 0);
#endif
}

// ---------------- fp32 -> bf16 conversion ----------------
__global__ void smha_cvt_bf16(const float* __restrict__ in, bf16* __restrict__ out, int n) {
  int i = blockIdx.x * blockDim.x + threadIdx.x;
  if (i < n) out[i] = (bf16)in[i];
}

// ---------------- fp32 -> bf16 transposed convert (1024x1024 weights) ------
__global__ __launch_bounds__(256)
void smha_cvt_bf16_T(const float* __restrict__ in, bf16* __restrict__ out) {
  __shared__ float tile[16][17];
  const int bx = blockIdx.x & 63, by = blockIdx.x >> 6;
  const int tx = threadIdx.x & 15, ty = threadIdx.x >> 4;
  tile[ty][tx] = in[(size_t)(by * 16 + ty) * E_ + bx * 16 + tx];
  __syncthreads();
  out[(size_t)(bx * 16 + ty) * E_ + by * 16 + tx] = (bf16)tile[tx][ty];
}

// ---------------- projection GEMM: Y = X @ W + b  (Wt is W transposed) -----
__global__ __launch_bounds__(32)
void smha_proj_gemm(const bf16* __restrict__ X, const bf16* __restrict__ Wt,
                    const float* __restrict__ bias, bf16* __restrict__ Y,
                    int transpose_out) {
  const int tn = blockIdx.x & 15;
  const int tm = blockIdx.x >> 4;
  const int m0 = tm * 16, n0 = tn * 64;
  const int lane = threadIdx.x;
  const int lm = lane & 15;
  const int koff = (lane >= 16) ? 8 : 0;

  v8f acc[4] = {v8f{}, v8f{}, v8f{}, v8f{}};
  const bf16* xrow = X + (size_t)(m0 + lm) * E_;
  const bf16* wcol[4];
#pragma unroll
  for (int t = 0; t < 4; t++) wcol[t] = Wt + (size_t)(n0 + t * 16 + lm) * E_;

  for (int k0 = 0; k0 < E_; k0 += 32) {
    v16bf a = frag_contig(xrow + k0, koff);
#pragma unroll
    for (int t = 0; t < 4; t++)
      acc[t] = wmma_bf16(a, frag_contig(wcol[t] + k0, koff), acc[t]);
  }

#pragma unroll
  for (int t = 0; t < 4; t++) {
#pragma unroll
    for (int r = 0; r < 8; r++) {
      int m = m0 + ((lane >= 16) ? 8 : 0) + r;
      int n = n0 + t * 16 + lm;
      int b = m >> 10, c = m & (C_ - 1);
      int h = n >> 6, dd = n & (D_ - 1);
      float val = acc[t][r] + bias[n];
      size_t idx = transpose_out
          ? (((size_t)b * H_ + h) * D_ + dd) * C_ + c
          : (((size_t)b * H_ + h) * C_ + c) * D_ + dd;
      Y[idx] = (bf16)val;
    }
  }
}

// ---------------- score_zero[b,h,c] = scale * <q_u, k_u> ----------------
__global__ void smha_score_zero(const bf16* __restrict__ Qu, const bf16* __restrict__ Ku,
                                float* __restrict__ sz) {
  int row = blockIdx.x * blockDim.x + threadIdx.x;
  if (row >= B_ * H_ * C_) return;
  const bf16* q = Qu + (size_t)row * D_;
  const bf16* k = Ku + (size_t)row * D_;
  float acc = 0.f;
#pragma unroll
  for (int d = 0; d < D_; d++) acc += (float)q[d] * (float)k[d];
  sz[row] = acc * SCALE_;
}

// ---------------- fused dual-softmax attention ----------------
// grid = B*H*(C/16); one wave per 16-query tile of one head.
// K row-major per key; Vt transposed (Vt[bh][d][c]).
__global__ __launch_bounds__(32)
void smha_attn(const bf16* __restrict__ Qu, const bf16* __restrict__ K,
               const bf16* __restrict__ Vt, const bf16* __restrict__ Vu,
               const float* __restrict__ sz, const float* __restrict__ mask,
               float* __restrict__ out0, float* __restrict__ out1) {
  extern __shared__ char smem[];                  // dynamic: LDS_TOTAL bytes
  float* sc = (float*)(smem + LDS_SC);
  bf16* pb = (bf16*)(smem + LDS_SC);

  const int qt = blockIdx.x & 63;
  const int bh = blockIdx.x >> 6;
  const int b = bh >> 4, h = bh & 15;
  const size_t head = (size_t)bh * C_ * D_;
  const int lane = threadIdx.x;
  const int lm = lane & 15;
  const int koff = (lane >= 16) ? 8 : 0;
  const int mb = (lane >= 16) ? 8 : 0;

  // ---- phase 1: S = scale * Qu . K^T via WMMA, K streamed by TDM ----
  const bf16* qrow = Qu + head + (size_t)(qt * 16 + lm) * D_;
  v16bf qa0 = frag_contig(qrow, koff);
  v16bf qa1 = frag_contig(qrow + 32, koff);

  // chunk = 256 keys (256x64 bf16 contiguous = 32KB), double-buffered
  tdm_load_2d_bf16(LDS_BUF0, K + head, D_, 256, D_, C_, D_);
  for (int ck = 0; ck < 4; ck++) {
    if (ck < 3) {
      tdm_load_2d_bf16((ck & 1) ? LDS_BUF0 : LDS_BUF1,
                       K + head + (size_t)(ck + 1) * 256 * D_, D_, 256, D_, C_, D_);
      __builtin_amdgcn_s_wait_tensorcnt(1);
    } else {
      __builtin_amdgcn_s_wait_tensorcnt(0);
    }
    const bf16* kbuf = (const bf16*)(smem + ((ck & 1) ? LDS_BUF1 : LDS_BUF0));
    for (int kt2 = 0; kt2 < 16; kt2++) {
      const bf16* krow = kbuf + (size_t)(kt2 * 16 + lm) * D_;
      v16bf kb0 = frag_contig(krow, koff);        // B[d][j] = K[j][d]
      v16bf kb1 = frag_contig(krow + 32, koff);
      v8f s = v8f{};
      s = wmma_bf16(qa0, kb0, s);
      s = wmma_bf16(qa1, kb1, s);
      int n = ck * 256 + kt2 * 16 + lm;
#pragma unroll
      for (int r = 0; r < 8; r++) sc[(mb + r) * C_ + n] = s[r] * SCALE_;
    }
  }
  __syncthreads();

  // ---- phase 2: dual softmax (normal + diag-substituted) ----
  float darr[16];
#pragma unroll
  for (int m = 0; m < 16; m++) {
    const int q = qt * 16 + m;
    const float* scr = sc + m * C_;
    const float* mrow = mask + (size_t)q * C_;
    const float sz_q = sz[(size_t)bh * C_ + q];
    const float dzu = sz_q + mrow[q];
    const int j0 = lane * 32;

    float v_[32];
#pragma unroll
    for (int t = 0; t < 32; t++) v_[t] = scr[j0 + t] + mrow[j0 + t];

    float m1 = -3.0e38f, m2p = -3.0e38f;
#pragma unroll
    for (int t = 0; t < 32; t++) {
      m1 = fmaxf(m1, v_[t]);
      m2p = fmaxf(m2p, (j0 + t == q) ? -3.0e38f : v_[t]);
    }
    m1 = wred_max(m1);
    const float m2 = fmaxf(wred_max(m2p), dzu);

    float s1 = 0.f, s2 = 0.f;
#pragma unroll
    for (int t = 0; t < 32; t++) {
      float e = __expf(v_[t] - m1);
      v_[t] = e;
      s1 += e;
      s2 += (j0 + t == q) ? 0.f : e;
    }
    s1 = wred_sum(s1);
    s2 = wred_sum(s2);
    const float f12 = __expf(m1 - m2);
    const float ed = __expf(dzu - m2);
    const float il1 = 1.f / s1;
    const float il2 = 1.f / (s2 * f12 + ed);
    darr[m] = ed * il2;

    bf16* p1r = pb + (size_t)m * 2048;             // overlays sc row m (regs staged)
    bf16* p2r = p1r + 1024;
#pragma unroll
    for (int t = 0; t < 32; t++) {
      int j = j0 + t;
      p1r[j] = (bf16)(v_[t] * il1);
      p2r[j] = (bf16)((j == q) ? 0.f : v_[t] * f12 * il2);
    }
  }
  __syncthreads();

  // ---- phase 3: O = P1.V, Ou = P2.V via WMMA, Vt streamed by TDM ----
  v8f o[4]  = {v8f{}, v8f{}, v8f{}, v8f{}};
  v8f ou[4] = {v8f{}, v8f{}, v8f{}, v8f{}};
  const bf16* p1row = pb + (size_t)lm * 2048;
  const bf16* p2row = p1row + 1024;

  // chunk = 64 d-rows x 256 keys (strided tile, row stride C_), double-buffered
  tdm_load_2d_bf16(LDS_BUF0, Vt + head, 256, D_, C_, D_, C_);
  for (int ck = 0; ck < 4; ck++) {
    if (ck < 3) {
      tdm_load_2d_bf16((ck & 1) ? LDS_BUF0 : LDS_BUF1,
                       Vt + head + (size_t)(ck + 1) * 256, 256, D_, C_, D_, C_);
      __builtin_amdgcn_s_wait_tensorcnt(1);
    } else {
      __builtin_amdgcn_s_wait_tensorcnt(0);
    }
    const bf16* vbuf = (const bf16*)(smem + ((ck & 1) ? LDS_BUF1 : LDS_BUF0));
    for (int c0 = 0; c0 < 256; c0 += 32) {
      v16bf a1 = frag_contig(p1row + ck * 256 + c0, koff);
      v16bf a2 = frag_contig(p2row + ck * 256 + c0, koff);
#pragma unroll
      for (int t = 0; t < 4; t++) {
        v16bf bv = frag_contig(vbuf + (size_t)(t * 16 + lm) * 256 + c0, koff);
        o[t]  = wmma_bf16(a1, bv, o[t]);
        ou[t] = wmma_bf16(a2, bv, ou[t]);
      }
    }
  }

  // ---- epilogue: add diag_attn * v_u, write merged-head fp32 outputs ----
#pragma unroll
  for (int t = 0; t < 4; t++) {
#pragma unroll
    for (int r = 0; r < 8; r++) {
      int m = mb + r;
      int q = qt * 16 + m;
      int n = t * 16 + lm;
      float da = (lane >= 16) ? darr[r + 8] : darr[r];
      float vu = (float)Vu[head + (size_t)q * D_ + n];
      size_t oi = ((size_t)b * C_ + q) * E_ + (size_t)h * D_ + n;
      out0[oi] = o[t][r];
      out1[oi] = ou[t][r] + da * vu;
    }
  }
}

// ---------------------------------------------------------------------------
extern "C" void kernel_launch(void* const* d_in, const int* in_sizes, int n_in,
                              void* d_out, int out_size, void* d_ws, size_t ws_size,
                              hipStream_t stream) {
  const float* embed   = (const float*)d_in[0];
  const float* embed_u = (const float*)d_in[1];
  const float* mask    = (const float*)d_in[2];
  const float* Wk      = (const float*)d_in[3];
  const float* bk      = (const float*)d_in[4];
  const float* Wq      = (const float*)d_in[5];
  const float* bq      = (const float*)d_in[6];
  const float* Wv      = (const float*)d_in[7];
  const float* bv      = (const float*)d_in[8];

  float* out0 = (float*)d_out;
  float* out1 = out0 + (size_t)B_ * C_ * E_;

  char* ws = (char*)d_ws;
  size_t off = 0;
  auto alloc = [&](size_t bytes) -> void* {
    void* p = ws + off;
    off += (bytes + 255) & ~(size_t)255;
    return p;
  };
  const size_t NE = (size_t)B_ * C_ * E_;
  const size_t NW = (size_t)E_ * E_;
  bf16* e_bf  = (bf16*)alloc(NE * 2);
  bf16* eu_bf = (bf16*)alloc(NE * 2);
  bf16* wkT = (bf16*)alloc(NW * 2);
  bf16* wqT = (bf16*)alloc(NW * 2);
  bf16* wvT = (bf16*)alloc(NW * 2);
  bf16* Kp  = (bf16*)alloc(NE * 2);           // head-major (B,H,C,D)
  bf16* Kup = (bf16*)alloc(NE * 2);
  bf16* Qup = (bf16*)alloc(NE * 2);
  bf16* Vtp = (bf16*)alloc(NE * 2);           // transposed (B,H,D,C)
  bf16* Vup = (bf16*)alloc(NE * 2);
  float* szp = (float*)alloc((size_t)B_ * H_ * C_ * 4);

  // 1) bf16 conversions (weights transposed for contiguous B-fragments)
  smha_cvt_bf16<<<(int)(NE / 256), 256, 0, stream>>>(embed,   e_bf,  (int)NE);
  smha_cvt_bf16<<<(int)(NE / 256), 256, 0, stream>>>(embed_u, eu_bf, (int)NE);
  smha_cvt_bf16_T<<<64 * 64, 256, 0, stream>>>(Wk, wkT);
  smha_cvt_bf16_T<<<64 * 64, 256, 0, stream>>>(Wq, wqT);
  smha_cvt_bf16_T<<<64 * 64, 256, 0, stream>>>(Wv, wvT);

  // 2) projections
  const int pg = (B_ * C_ / 16) * (E_ / 64);
  smha_proj_gemm<<<pg, 32, 0, stream>>>(e_bf,  wkT, bk, Kp,  0);
  smha_proj_gemm<<<pg, 32, 0, stream>>>(eu_bf, wkT, bk, Kup, 0);
  smha_proj_gemm<<<pg, 32, 0, stream>>>(eu_bf, wqT, bq, Qup, 0);
  smha_proj_gemm<<<pg, 32, 0, stream>>>(e_bf,  wvT, bv, Vtp, 1);
  smha_proj_gemm<<<pg, 32, 0, stream>>>(eu_bf, wvT, bv, Vup, 0);

  // 3) score_zero
  smha_score_zero<<<(B_ * H_ * C_) / 256, 256, 0, stream>>>(Qup, Kup, szp);

  // 4) fused dual-softmax attention (128KB dynamic LDS for TDM + scores)
  smha_attn<<<B_ * H_ * (C_ / 16), 32, LDS_TOTAL, stream>>>(Qup, Kp, Vtp, Vup,
                                                            szp, mask, out0, out1);
  (void)in_sizes; (void)n_in; (void)out_size; (void)ws_size;
}
